// DrugGAT_13864154431797
// MI455X (gfx1250) — matrix operand order repaired and checked by reading, not compiled, not run on hardware
//
#include <hip/hip_runtime.h>

typedef __attribute__((ext_vector_type(16))) _Float16 v16h;
typedef __attribute__((ext_vector_type(8)))  float    v8f;
typedef __attribute__((ext_vector_type(4)))  int      v4i;

#define NEG_SLOPE 0.2f
#define BN_EPS    1e-5f

#define MBLK  256
#define NBLK  64
#define KSTEP 32

#if defined(__HIP_DEVICE_COMPILE__) && __has_builtin(__builtin_amdgcn_global_load_async_to_lds_b128)
#define USE_ASYNC_LDS 1
#else
#define USE_ASYNC_LDS 0
#endif

// ---------- 16-byte global -> LDS copy (async on CDNA5 if available) ----------
__device__ __forceinline__ void cp16(const _Float16* __restrict__ g, _Float16* l) {
#if USE_ASYNC_LDS
  __builtin_amdgcn_global_load_async_to_lds_b128(
      (__attribute__((address_space(1))) v4i*)(v4i*)(void*)g,
      (__attribute__((address_space(3))) v4i*)(v4i*)(void*)l, 0, 0);
#else
  *(float4*)l = *(const float4*)g;
#endif
}

__device__ __forceinline__ void async_fence() {
#if USE_ASYNC_LDS
#if __has_builtin(__builtin_amdgcn_s_wait_asynccnt)
  __builtin_amdgcn_s_wait_asynccnt(0);
#else
  asm volatile("s_wait_asynccnt 0x0" ::: "memory");
#endif
#endif
}

// ---------- float <-> ordered-uint for atomic float max ----------
__device__ __forceinline__ unsigned f2ord(float f) {
  unsigned u = __float_as_uint(f);
  return (u & 0x80000000u) ? ~u : (u | 0x80000000u);
}
__device__ __forceinline__ float ord2f(unsigned u) {
  return (u & 0x80000000u) ? __uint_as_float(u & 0x7fffffffu)
                           : __uint_as_float(~u);
}

// ---------- fp32 -> fp16 convert (activations, row-major) ----------
__global__ void k_f32_to_f16(const float* __restrict__ in,
                             _Float16* __restrict__ out, int n) {
  int i = blockIdx.x * blockDim.x + threadIdx.x;
  if (i < n) out[i] = (_Float16)in[i];
}

// ---------- fp32 -> fp16 convert + transpose (weights: W[K,Nn] -> WT[Nn,K]) ----------
__global__ void k_w_to_f16_t(const float* __restrict__ w, _Float16* __restrict__ wt,
                             int K, int Nn) {
  int i = blockIdx.x * blockDim.x + threadIdx.x;
  if (i >= K * Nn) return;
  int k = i / Nn, n = i - k * Nn;
  wt[(long long)n * K + k] = (_Float16)w[i];
}

// ---------- WMMA GEMM: C[M,Nn] = A[M,K](f16) * WT[Nn,K](f16, pre-transposed) ----------
// 256 threads = 8 waves; wave w owns rows [32w,32w+32) of the 256x64 block tile
// => 2 A-frags x 4 B-frags = 8 WMMAs per K-step. Double-buffered LDS.
__global__ __launch_bounds__(256)
void k_gemm_f16(const _Float16* __restrict__ A, const _Float16* __restrict__ WT,
                float* __restrict__ C, int M, int K, int Nn) {
  __shared__ __align__(16) _Float16 sA[2][MBLK][KSTEP];  // 2 x 16 KB
  __shared__ __align__(16) _Float16 sB[2][NBLK][KSTEP];  // 2 x 4 KB
  const int tid  = threadIdx.x;
  const int wave = tid >> 5;
  const int lane = tid & 31;
  const int m0   = blockIdx.x * MBLK;
  const int n0   = blockIdx.y * NBLK;
  const bool full = (m0 + MBLK) <= M;

  union Acc { v8f v; float f[8]; } acc[2][4];
  #pragma unroll
  for (int mt = 0; mt < 2; ++mt)
    #pragma unroll
    for (int nt = 0; nt < 4; ++nt)
      #pragma unroll
      for (int j = 0; j < 8; ++j) acc[mt][nt].f[j] = 0.f;

  // ---- tile staging (cooperative, 16B granules) ----
  auto stage = [&](int buf, int k0) {
    // A: 256x32 f16 = 1024 granules, 4 per thread
    #pragma unroll
    for (int ii = 0; ii < (MBLK * KSTEP / 8) / 256; ++ii) {
      int i = ii * 256 + tid;
      int r = i >> 2;            // row 0..255
      int c = (i & 3) * 8;       // 0,8,16,24
      if (full || (m0 + r < M))
        cp16(A + (long long)(m0 + r) * K + k0 + c, &sA[buf][r][c]);
      else
        *(float4*)&sA[buf][r][c] = make_float4(0.f, 0.f, 0.f, 0.f);
    }
    // B (already transposed in memory): 64x32 f16 = 256 granules, 1 per thread
    {
      int r = tid >> 2;          // 0..63 (n within tile)
      int c = (tid & 3) * 8;     // 0,8,16,24 (k within tile)
      cp16(WT + (long long)(n0 + r) * K + k0 + c, &sB[buf][r][c]);
    }
  };

  stage(0, 0);
  async_fence();
  __syncthreads();

  const int am  = lane & 15;
  const int kb  = (lane < 16) ? 0 : 8;   // A frag K-base (16-bit A 16x32 layout)
  const int kb2 = (lane < 16) ? 0 : 16;  // unused: B frags are read with kb, same k order as A

  for (int k0 = 0; k0 < K; k0 += KSTEP) {
    const int buf = (k0 / KSTEP) & 1;
    const bool more = (k0 + KSTEP) < K;
    if (more) stage(buf ^ 1, k0 + KSTEP);

    union AF { v16h v; unsigned u[8]; } af[2];
    #pragma unroll
    for (int mt = 0; mt < 2; ++mt) {
      int row = wave * 32 + mt * 16 + am;
      #pragma unroll
      for (int q = 0; q < 8; ++q) {
        int kk = ((q < 4) ? 0 : 16) + kb + 2 * (q & 3);
        af[mt].u[q] = *(const unsigned*)&sA[buf][row][kk];
      }
    }
    #pragma unroll
    for (int nt = 0; nt < 4; ++nt) {
      union BF { v16h v; unsigned u[8]; } bf;
      int bn = nt * 16 + am;
      #pragma unroll
      for (int q = 0; q < 8; ++q)
        bf.u[q] = *(const unsigned*)&sB[buf][bn][((q < 4) ? 0 : 16) + kb + 2 * (q & 3)];
      #pragma unroll
      for (int mt = 0; mt < 2; ++mt)
        acc[mt][nt].v = __builtin_amdgcn_wmma_f32_16x16x32_f16(
            false, af[mt].v, false, bf.v, (short)0, acc[mt][nt].v, false, false);
    }
    if (more) async_fence();
    __syncthreads();
  }

  // ---- store C (16x16 f32 C/D layout: VGPR r -> M=r / M=8+r) ----
  int col    = lane & 15;
  int rowoff = (lane < 16) ? 0 : 8;
  #pragma unroll
  for (int mt = 0; mt < 2; ++mt)
    #pragma unroll
    for (int nt = 0; nt < 4; ++nt)
      #pragma unroll
      for (int r = 0; r < 8; ++r) {
        int row = m0 + wave * 32 + mt * 16 + rowoff + r;
        if (full || row < M)
          C[(long long)row * Nn + n0 + nt * 16 + col] = acc[mt][nt].f[r];
      }
}

// ---------- edge pass 1: e = att . leaky_relu(xl[src]+xr[dst]); segment max ----------
__global__ void k_edge_logits(const float* __restrict__ xl, const float* __restrict__ xr,
                              const float* __restrict__ att, const long long* __restrict__ ei,
                              int E, int N, int H, int F,
                              float* __restrict__ e, unsigned* __restrict__ mmax) {
  int t = blockIdx.x * blockDim.x + threadIdx.x;
  int total = (E + N) * H;
  if (t >= total) return;
  int eid = t / H, h = t - eid * H;
  long long s, d;
  if (eid < E) { s = ei[eid]; d = ei[E + eid]; } else { s = d = eid - E; }
  const int C = H * F;
  const float* pl = xl + s * (long long)C + h * F;
  const float* pr = xr + d * (long long)C + h * F;
  const float* pa = att + h * F;
  float acc = 0.f;
  for (int f = 0; f < F; f += 4) {
    float4 a = *(const float4*)(pl + f);
    float4 b = *(const float4*)(pr + f);
    float4 w = *(const float4*)(pa + f);
    float v;
    v = a.x + b.x; v = v > 0.f ? v : NEG_SLOPE * v; acc += v * w.x;
    v = a.y + b.y; v = v > 0.f ? v : NEG_SLOPE * v; acc += v * w.y;
    v = a.z + b.z; v = v > 0.f ? v : NEG_SLOPE * v; acc += v * w.z;
    v = a.w + b.w; v = v > 0.f ? v : NEG_SLOPE * v; acc += v * w.w;
  }
  e[t] = acc;
  atomicMax(mmax + d * H + h, f2ord(acc));
}

// ---------- edge pass 2: ex = exp(e - m[dst]); segment sum of ex ----------
__global__ void k_edge_exp(float* __restrict__ e, const long long* __restrict__ ei,
                           int E, int N, int H,
                           const unsigned* __restrict__ mmax, float* __restrict__ den) {
  int t = blockIdx.x * blockDim.x + threadIdx.x;
  int total = (E + N) * H;
  if (t >= total) return;
  int eid = t / H, h = t - eid * H;
  long long d;
  if (eid < E) d = ei[E + eid]; else d = eid - E;
  float ex = __expf(e[t] - ord2f(mmax[d * H + h]));
  e[t] = ex;
  atomicAdd(den + d * H + h, ex);
}

// ---------- edge pass 3: acc[dst] += ex * xl[src]  (alpha-division hoisted) ----------
__global__ void k_edge_scatter(const float* __restrict__ e, const float* __restrict__ xl,
                               const long long* __restrict__ ei,
                               int E, int N, int H, int F, float* __restrict__ accb) {
  int t = blockIdx.x * blockDim.x + threadIdx.x;
  int total = (E + N) * H;
  if (t >= total) return;
  int eid = t / H, h = t - eid * H;
  long long s, d;
  if (eid < E) { s = ei[eid]; d = ei[E + eid]; } else { s = d = eid - E; }
  const int C = H * F;
  float ex = e[t];
  const float* pl = xl + s * (long long)C + h * F;
  float* pd = accb + d * (long long)C + h * F;
  for (int f = 0; f < F; f += 4) {
    float4 a = *(const float4*)(pl + f);
    atomicAdd(pd + f + 0, ex * a.x);
    atomicAdd(pd + f + 1, ex * a.y);
    atomicAdd(pd + f + 2, ex * a.z);
    atomicAdd(pd + f + 3, ex * a.w);
  }
}

// ---------- node finish (layers 1,2): /den + bias, BN, ELU, emit f16 ----------
__global__ void k_node_bn_elu(const float* __restrict__ accb, const float* __restrict__ den,
                              const float* __restrict__ bias,
                              const float* __restrict__ g, const float* __restrict__ bb,
                              const float* __restrict__ rm, const float* __restrict__ rv,
                              _Float16* __restrict__ hout, int N, int H, int F) {
  int i = blockIdx.x * blockDim.x + threadIdx.x;
  int C = H * F;
  if (i >= N * C) return;
  int n = i / C, c = i - n * C, h = c / F;
  float v = accb[i] / den[n * H + h] + bias[c];
  v = (v - rm[c]) * rsqrtf(rv[c] + BN_EPS) * g[c] + bb[c];
  v = v > 0.f ? v : expm1f(v);
  hout[i] = (_Float16)v;
}

// ---------- node finish (layer 3, H=1): /den + bias, fp32 out ----------
__global__ void k_node_out(const float* __restrict__ accb, const float* __restrict__ den,
                           const float* __restrict__ bias, float* __restrict__ out,
                           int N, int C) {
  int i = blockIdx.x * blockDim.x + threadIdx.x;
  if (i >= N * C) return;
  int n = i / C, c = i - n * C;
  out[i] = accb[i] / den[n] + bias[c];
}

extern "C" void kernel_launch(void* const* d_in, const int* in_sizes, int n_in,
                              void* d_out, int out_size, void* d_ws, size_t ws_size,
                              hipStream_t stream) {
  const float*     x    = (const float*)d_in[0];
  const long long* ei   = (const long long*)d_in[1];  // int64 edge_index [2,E]
  const float* Wl1 = (const float*)d_in[2];
  const float* Wr1 = (const float*)d_in[3];
  const float* att1= (const float*)d_in[4];
  const float* b1  = (const float*)d_in[5];
  const float* g1f = (const float*)d_in[6];
  const float* b1f = (const float*)d_in[7];
  const float* m1f = (const float*)d_in[8];
  const float* v1f = (const float*)d_in[9];
  const float* Wl2 = (const float*)d_in[10];
  const float* Wr2 = (const float*)d_in[11];
  const float* att2= (const float*)d_in[12];
  const float* b2  = (const float*)d_in[13];
  const float* g2f = (const float*)d_in[14];
  const float* b2f = (const float*)d_in[15];
  const float* m2f = (const float*)d_in[16];
  const float* v2f = (const float*)d_in[17];
  const float* Wl3 = (const float*)d_in[18];
  const float* Wr3 = (const float*)d_in[19];
  const float* att3= (const float*)d_in[20];
  const float* b3  = (const float*)d_in[21];
  (void)n_in; (void)out_size; (void)ws_size;

  const int N = in_sizes[0] / 128;   // 50000
  const int E = in_sizes[1] / 2;     // 800000

  char* ws = (char*)d_ws;
  size_t off = 0;
  auto carve = [&](size_t bytes) -> char* {
    char* p = ws + off;
    off = (off + bytes + 255) & ~(size_t)255;
    return p;
  };
  _Float16* x16  = (_Float16*)carve((size_t)N * 256 * 2);  // f16 activations
  _Float16* w16a = (_Float16*)carve((size_t)256 * 256 * 2);  // transposed f16 weights
  _Float16* w16b = (_Float16*)carve((size_t)256 * 256 * 2);
  float*    xl   = (float*)carve((size_t)N * 256 * 4);
  float*    xr   = (float*)carve((size_t)N * 256 * 4);
  float*    accb = (float*)carve((size_t)N * 256 * 4);
  float*    ebuf = (float*)carve((size_t)(E + N) * 4 * 4);
  unsigned* mmax = (unsigned*)carve((size_t)N * 4 * 4);
  float*    den  = (float*)carve((size_t)N * 4 * 4);

  dim3 blk(256);
  auto cdiv = [](long long a, long long b) { return (unsigned)((a + b - 1) / b); };

  // ================= Layer 1: Cin=128 -> 256, H=4, F=64 =================
  k_f32_to_f16<<<cdiv((long long)N * 128, 256), blk, 0, stream>>>(x, x16, N * 128);
  k_w_to_f16_t<<<cdiv(128 * 256, 256), blk, 0, stream>>>(Wl1, w16a, 128, 256);
  k_w_to_f16_t<<<cdiv(128 * 256, 256), blk, 0, stream>>>(Wr1, w16b, 128, 256);
  {
    dim3 g(cdiv(N, MBLK), 256 / NBLK);
    k_gemm_f16<<<g, blk, 0, stream>>>(x16, w16a, xl, N, 128, 256);
    k_gemm_f16<<<g, blk, 0, stream>>>(x16, w16b, xr, N, 128, 256);
  }
  (void)hipMemsetAsync(accb, 0, (size_t)N * 256 * 4, stream);
  (void)hipMemsetAsync(mmax, 0, (size_t)N * 4 * 4, stream);
  (void)hipMemsetAsync(den,  0, (size_t)N * 4 * 4, stream);
  {
    int tot = (E + N) * 4;
    k_edge_logits<<<cdiv(tot, 256), blk, 0, stream>>>(xl, xr, att1, ei, E, N, 4, 64, ebuf, mmax);
    k_edge_exp   <<<cdiv(tot, 256), blk, 0, stream>>>(ebuf, ei, E, N, 4, mmax, den);
    k_edge_scatter<<<cdiv(tot, 256), blk, 0, stream>>>(ebuf, xl, ei, E, N, 4, 64, accb);
  }
  k_node_bn_elu<<<cdiv((long long)N * 256, 256), blk, 0, stream>>>(
      accb, den, b1, g1f, b1f, m1f, v1f, x16, N, 4, 64);

  // ================= Layer 2: 256 -> 256, H=4, F=64 =================
  k_w_to_f16_t<<<cdiv(256 * 256, 256), blk, 0, stream>>>(Wl2, w16a, 256, 256);
  k_w_to_f16_t<<<cdiv(256 * 256, 256), blk, 0, stream>>>(Wr2, w16b, 256, 256);
  {
    dim3 g(cdiv(N, MBLK), 256 / NBLK);
    k_gemm_f16<<<g, blk, 0, stream>>>(x16, w16a, xl, N, 256, 256);
    k_gemm_f16<<<g, blk, 0, stream>>>(x16, w16b, xr, N, 256, 256);
  }
  (void)hipMemsetAsync(accb, 0, (size_t)N * 256 * 4, stream);
  (void)hipMemsetAsync(mmax, 0, (size_t)N * 4 * 4, stream);
  (void)hipMemsetAsync(den,  0, (size_t)N * 4 * 4, stream);
  {
    int tot = (E + N) * 4;
    k_edge_logits<<<cdiv(tot, 256), blk, 0, stream>>>(xl, xr, att2, ei, E, N, 4, 64, ebuf, mmax);
    k_edge_exp   <<<cdiv(tot, 256), blk, 0, stream>>>(ebuf, ei, E, N, 4, mmax, den);
    k_edge_scatter<<<cdiv(tot, 256), blk, 0, stream>>>(ebuf, xl, ei, E, N, 4, 64, accb);
  }
  k_node_bn_elu<<<cdiv((long long)N * 256, 256), blk, 0, stream>>>(
      accb, den, b2, g2f, b2f, m2f, v2f, x16, N, 4, 64);

  // ================= Layer 3: 256 -> 128, H=1, F=128, mean==identity =====
  k_w_to_f16_t<<<cdiv(256 * 128, 256), blk, 0, stream>>>(Wl3, w16a, 256, 128);
  k_w_to_f16_t<<<cdiv(256 * 128, 256), blk, 0, stream>>>(Wr3, w16b, 256, 128);
  {
    dim3 g(cdiv(N, MBLK), 128 / NBLK);
    k_gemm_f16<<<g, blk, 0, stream>>>(x16, w16a, xl, N, 256, 128);
    k_gemm_f16<<<g, blk, 0, stream>>>(x16, w16b, xr, N, 256, 128);
  }
  (void)hipMemsetAsync(accb, 0, (size_t)N * 128 * 4, stream);
  (void)hipMemsetAsync(mmax, 0, (size_t)N * 1 * 4, stream);
  (void)hipMemsetAsync(den,  0, (size_t)N * 1 * 4, stream);
  {
    int tot = (E + N) * 1;
    k_edge_logits<<<cdiv(tot, 256), blk, 0, stream>>>(xl, xr, att3, ei, E, N, 1, 128, ebuf, mmax);
    k_edge_exp   <<<cdiv(tot, 256), blk, 0, stream>>>(ebuf, ei, E, N, 1, mmax, den);
    k_edge_scatter<<<cdiv(tot, 256), blk, 0, stream>>>(ebuf, xl, ei, E, N, 1, 128, accb);
  }
  k_node_out<<<cdiv((long long)N * 128, 256), blk, 0, stream>>>(
      accb, den, b3, (float*)d_out, N, 128);
}
